// ModConv_47261820125769
// MI455X (gfx1250) — matrix operand order repaired and checked by reading, not compiled, not run on hardware
//
#include <hip/hip_runtime.h>

typedef _Float16 half_t;
typedef __attribute__((ext_vector_type(16))) _Float16 v16h;
typedef __attribute__((ext_vector_type(8)))  _Float16 v8h;
typedef __attribute__((ext_vector_type(8)))  float    v8f;

#define LATN 512
#define FIN  256
#define FOUT 256
#define HWD  64
#define KTOT 2304   // 9 * FIN, layout k = r*256 + i (spatial tap outer, channel inner)

// ---------------- Style MLP: one EqualLinear + LeakyReLU(0.2) layer ----------------
__global__ __launch_bounds__(128) void mlp_layer(const float* __restrict__ hin,
                                                 float* __restrict__ hout,
                                                 const float* __restrict__ W,
                                                 const float* __restrict__ bias) {
    const int j = blockIdx.x * 128 + threadIdx.x;   // 512 output features
    const float eq = 0.044194173824159216f;         // 1/sqrt(512)
    float acc[16];
#pragma unroll
    for (int b = 0; b < 16; ++b) acc[b] = 0.f;
    const float* wrow = W + (size_t)j * LATN;
    for (int i = 0; i < LATN; ++i) {
        const float w = wrow[i];
#pragma unroll
        for (int b = 0; b < 16; ++b) acc[b] += hin[b * LATN + i] * w;
    }
    const float bj = bias[j];
#pragma unroll
    for (int b = 0; b < 16; ++b) {
        float v = acc[b] * eq + bj;
        hout[b * LATN + j] = (v > 0.f) ? v : 0.2f * v;
    }
}

// ---------------- Final linear (512->256) + per-sample inf-norm ----------------
__global__ __launch_bounds__(256) void style_final(const float* __restrict__ h,
                                                   const float* __restrict__ wf,
                                                   const float* __restrict__ bfv,
                                                   float* __restrict__ sn) {
    __shared__ float s[16 * 256];
    __shared__ float mx[16];
    const int j = threadIdx.x;                      // 256 output features
    float acc[16];
#pragma unroll
    for (int b = 0; b < 16; ++b) acc[b] = 0.f;
    const float* wrow = wf + (size_t)j * LATN;
    for (int i = 0; i < LATN; ++i) {
        const float w = wrow[i];
#pragma unroll
        for (int b = 0; b < 16; ++b) acc[b] += h[b * LATN + i] * w;
    }
    const float bj = bfv[j];
#pragma unroll
    for (int b = 0; b < 16; ++b) s[b * 256 + j] = acc[b] + bj;
    __syncthreads();
    if (j < 16) {
        float m = 0.f;
        for (int t = 0; t < 256; ++t) m = fmaxf(m, fabsf(s[j * 256 + t]));
        mx[j] = m;
    }
    __syncthreads();
#pragma unroll
    for (int b = 0; b < 16; ++b) sn[b * 256 + j] = s[b * 256 + j] / mx[b];
}

// ---------------- Modulate + demodulate, write f16 weights in k = r*256+i ----------------
__global__ __launch_bounds__(256) void modweight(const float* __restrict__ wt,
                                                 const float* __restrict__ sn,
                                                 half_t* __restrict__ wmod) {
    __shared__ float red[256];
    const int bo = blockIdx.x;
    const int b  = bo >> 8;
    const int o  = bo & 255;
    const int i  = threadIdx.x;                     // input channel (FIN == 256)

    float w9[9];
    const float* src = wt + ((size_t)o * FIN + i) * 9;
    float m = 0.f;
#pragma unroll
    for (int r = 0; r < 9; ++r) { w9[r] = src[r]; m = fmaxf(m, fabsf(w9[r])); }

    red[i] = m; __syncthreads();
    for (int s = 128; s > 0; s >>= 1) { if (i < s) red[i] = fmaxf(red[i], red[i + s]); __syncthreads(); }
    const float wmax = red[0];
    __syncthreads();

    const float scale = (1.0f / 48.0f) / wmax;      // 1/sqrt(FIN*9) = 1/48
    const float sv = sn[b * 256 + i] * scale;
    float ss = 0.f;
#pragma unroll
    for (int r = 0; r < 9; ++r) { w9[r] *= sv; ss += w9[r] * w9[r]; }

    red[i] = ss; __syncthreads();
    for (int s = 128; s > 0; s >>= 1) { if (i < s) red[i] += red[i + s]; __syncthreads(); }
    const float d = rsqrtf(red[0] + 1e-8f);

    half_t* dst = wmod + ((size_t)(b * FOUT + o)) * KTOT + i;
#pragma unroll
    for (int r = 0; r < 9; ++r) dst[r * 256] = (half_t)(w9[r] * d);
}

// ---------------- Implicit-GEMM conv via V_WMMA_F32_16X16X32_F16 ----------------
// Workgroup: 128 fout x 128 pixels (2 image rows). 8 waves as 4(M) x 2(N),
// wave tile 32x64 = 2x4 WMMA tiles. Double-buffered LDS (40 KB), pitch 40 halves
// (80 B: 16B-aligned, 20-bank stagger -> conflict-free b128 fragment loads).
__global__ __launch_bounds__(256) void conv_wmma(const float* __restrict__ x,
                                                 const half_t* __restrict__ wmod,
                                                 float* __restrict__ out) {
    __shared__ __align__(16) half_t As[2][128 * 40];   // [buf][fout_row][k]
    __shared__ __align__(16) half_t Bs[2][128 * 40];   // [buf][pixel][k]

    const int ntile = blockIdx.x;                   // 0..31  pixel tiles
    const int mtile = blockIdx.y;                   // 0..1   fout tiles
    const int b     = blockIdx.z;                   // 0..15  sample
    const int tid   = threadIdx.x;
    const int lane  = tid & 31;
    const int wv    = tid >> 5;
    const int mw    = wv & 3;                       // wave M index (0..3)
    const int nw    = wv >> 2;                      // wave N index (0..1)
    const int obase = mtile * 128;
    const int pbase = ntile * 128;
    const int h0    = ntile * 2;

    v8f acc[2][4];
#pragma unroll
    for (int a = 0; a < 2; ++a)
#pragma unroll
        for (int c = 0; c < 4; ++c)
#pragma unroll
            for (int e = 0; e < 8; ++e) acc[a][c][e] = 0.f;

    const half_t* wbase = wmod + ((size_t)(b * FOUT + obase)) * KTOT;
    const float*  xbase = x + (size_t)b * FIN * HWD * HWD;

    // --- per-thread staging constants ---
    const int arow = tid >> 1;                      // A staging: row 0..127
    const int acol = (tid & 1) * 16;                // 16 halves per thread chunk
    const int p    = tid & 127;                     // B staging pixel (constant per thread!)
    const int kk0  = tid >> 7;                      // 0/1: k parity; kk = 2*j + kk0
    const int prow = p >> 6;                        // image row within tile
    const int pw   = p & 63;                        // image column
    const int bofs = p * 40 + kk0;                  // LDS base for B stores

    // staging registers
    uint4 a0, a1;
    float bv[16];

    auto stage_issue = [&](int kb) {
        const uint4* s = (const uint4*)(wbase + (size_t)arow * KTOT + kb + acol);
        a0 = s[0];
        a1 = s[1];
        const int r  = kb >> 8;                     // spatial tap (K slice stays inside one tap)
        const int i0 = kb & 255;
        const int dh = r / 3 - 1;
        const int dw = r % 3 - 1;
        const int hh = h0 + prow + dh;
        const int ww = pw + dw;
        const bool ok = ((unsigned)hh < 64u) && ((unsigned)ww < 64u);
        const float* xs = xbase + ((size_t)(i0 + kk0) * HWD + hh) * HWD + ww;
#pragma unroll
        for (int j = 0; j < 16; ++j) {              // fixed stride: 2 channels apart
            bv[j] = 0.f;
            if (ok) bv[j] = xs[(size_t)j * 2 * HWD * HWD];
        }
    };
    auto stage_write = [&](int buf) {
        uint4* d = (uint4*)(&As[buf][arow * 40 + acol]);
        d[0] = a0;
        d[1] = a1;
#pragma unroll
        for (int j = 0; j < 16; ++j)
            Bs[buf][bofs + 2 * j] = (half_t)bv[j];  // b16 stores, immediate offsets
    };

    const int NT = KTOT / 32;                       // 72 K-steps
    stage_issue(0);
    stage_write(0);
    __syncthreads();

    for (int kc = 0; kc < NT; ++kc) {
        const int cur = kc & 1;
        const bool more = (kc + 1) < NT;
        if (more) stage_issue((kc + 1) * 32);       // global loads in flight during WMMAs
        if (kc + 2 < NT)                            // prefetch weight stream (global_prefetch_b8)
            __builtin_prefetch(wbase + (size_t)arow * KTOT + (kc + 2) * 32 + acol, 0, 3);

        // ---- compute on buffer `cur` ----
        v16h af[2], bfr[4];
        const int hsel = (lane >> 4) * 8;           // A: lanes<16 K{0..7,16..23}, else K{8..15,24..31}
#pragma unroll
        for (int mi = 0; mi < 2; ++mi) {
            const int row = mw * 32 + mi * 16 + (lane & 15);
            v8h lo = *(const v8h*)(&As[cur][row * 40 + hsel]);
            v8h hi = *(const v8h*)(&As[cur][row * 40 + 16 + hsel]);
            af[mi] = __builtin_shufflevector(lo, hi, 0,1,2,3,4,5,6,7,8,9,10,11,12,13,14,15);
        }
        const int ksel = (lane >> 4) * 8;           // B: lanes<16 K{0..7,16..23}, else K{8..15,24..31}; N = lane%16
#pragma unroll
        for (int ni = 0; ni < 4; ++ni) {
            const int prw = nw * 64 + ni * 16 + (lane & 15);
            v8h lo = *(const v8h*)(&Bs[cur][prw * 40 + ksel]);
            v8h hi = *(const v8h*)(&Bs[cur][prw * 40 + ksel + 16]);
            bfr[ni] = __builtin_shufflevector(lo, hi, 0,1,2,3,4,5,6,7,8,9,10,11,12,13,14,15);
        }
#pragma unroll
        for (int mi = 0; mi < 2; ++mi)
#pragma unroll
            for (int ni = 0; ni < 4; ++ni)
                acc[mi][ni] = __builtin_amdgcn_wmma_f32_16x16x32_f16(
                    false, af[mi], false, bfr[ni], (short)0, acc[mi][ni], false, false);

        if (more) stage_write(cur ^ 1);             // write next buffer after compute
        __syncthreads();
    }

    // Epilogue: C layout -> lanes<16: M=v,N=lane ; lanes>=16: M=v+8,N=lane-16
    const int ncol = lane & 15;
    const int orow = (lane >> 4) << 3;
#pragma unroll
    for (int mi = 0; mi < 2; ++mi) {
        const int og = obase + mw * 32 + mi * 16 + orow;
#pragma unroll
        for (int ni = 0; ni < 4; ++ni) {
            const int pg = pbase + nw * 64 + ni * 16 + ncol;
            float* dst = out + ((size_t)(b * FOUT + og)) * 4096 + pg;
            v8f c = acc[mi][ni];
#pragma unroll
            for (int v = 0; v < 8; ++v) dst[(size_t)v * 4096] = c[v];
        }
    }
}

extern "C" void kernel_launch(void* const* d_in, const int* in_sizes, int n_in,
                              void* d_out, int out_size, void* d_ws, size_t ws_size,
                              hipStream_t stream) {
    const float* x   = (const float*)d_in[0];   // [16,256,64,64]
    const float* lat = (const float*)d_in[1];   // [16,512]
    const float* wt  = (const float*)d_in[2];   // [256,256,3,3]
    const float* wsS = (const float*)d_in[3];   // [4,512,512]
    const float* bsS = (const float*)d_in[4];   // [4,512]
    const float* wf  = (const float*)d_in[5];   // [256,512]
    const float* bfv = (const float*)d_in[6];   // [256]
    float* out = (float*)d_out;

    // Workspace carve (needs ~19 MB): hA | hB | sn | wmod(f16)
    float* hA = (float*)d_ws;
    float* hB = hA + 16 * LATN;
    float* sn = hB + 16 * LATN;
    half_t* wmod = (half_t*)(sn + 16 * FIN);

    mlp_layer<<<4, 128, 0, stream>>>(lat, hA, wsS + 0 * 262144, bsS + 0 * 512);
    mlp_layer<<<4, 128, 0, stream>>>(hA, hB, wsS + 1 * 262144, bsS + 1 * 512);
    mlp_layer<<<4, 128, 0, stream>>>(hB, hA, wsS + 2 * 262144, bsS + 2 * 512);
    mlp_layer<<<4, 128, 0, stream>>>(hA, hB, wsS + 3 * 262144, bsS + 3 * 512);
    style_final<<<1, 256, 0, stream>>>(hB, wf, bfv, sn);
    modweight<<<16 * 256, 256, 0, stream>>>(wt, sn, wmod);

    dim3 grid(32, 2, 16);   // pixel tiles x fout tiles x batch
    conv_wmma<<<grid, 256, 0, stream>>>(x, wmod, out);
}
